// CurricularFace_56710748176753
// MI455X (gfx1250) — matrix operand order repaired and checked by reading, not compiled, not run on hardware
//
#include <hip/hip_runtime.h>
#include <hip/hip_bf16.h>
#include <math.h>

#define NROWS 512
#define DDIM  512
#define NCLS  100000
#define NBX   782            // ceil(NCLS/128)
#define SCALEF 64.0f
#define EPSC  1e-7f
#define COS_M_F 0.8775825618903728f
#define SIN_M_F 0.479425538604203f
#define THRESH_F (-0.8775825618903728f)
#define MM_F 0.2397127693021015f

typedef __attribute__((ext_vector_type(16))) __bf16 v16bf;
typedef __attribute__((ext_vector_type(8)))  float  v8f;
typedef unsigned v4u __attribute__((ext_vector_type(4)));
typedef unsigned v8u __attribute__((ext_vector_type(8)));

static __device__ inline void splitbf(float f, __bf16& h, __bf16& l) {
    h = (__bf16)f;
    l = (__bf16)(f - (float)h);
}
static __device__ inline unsigned pack2(__bf16 a, __bf16 b) {
    union { __bf16 h[2]; unsigned u; } z; z.h[0] = a; z.h[1] = b; return z.u;
}
static __device__ inline unsigned short bfbits(__bf16 h) {
    union { __bf16 b; unsigned short u; } z; z.b = h; return z.u;
}

// A fragment: lane holds row m, K = {half*8 + 0..7} and {16 + half*8 + 0..7}  (ISA 7.12.2)
static __device__ inline v16bf ldA(const __bf16* p, int kc, int half) {
    union { v16bf v; uint4 q[2]; } u;
    u.q[0] = *(const uint4*)(p + kc + half * 8);
    u.q[1] = *(const uint4*)(p + kc + 16 + half * 8);
    return u.v;
}
// B fragment: lane holds column n, K = {half*8 + 0..7} and {16 + half*8 + 0..7}
static __device__ inline v16bf ldB(const __bf16* p, int kc, int half) {
    union { v16bf v; uint4 q[2]; } u;
    const __bf16* q = p + kc + half * 8;
    u.q[0] = *(const uint4*)(q);
    u.q[1] = *(const uint4*)(q + 16);
    return u.v;
}

// ---- Tensor Data Mover: 2D tile [128 rows x 64 bf16], row stride 512 elements,
// ---- LDS padding 4 DWORDs after every 32 DWORDs -> LDS row stride 72 bf16.
static __device__ inline void tdm_load_a_tile(unsigned lds_off, const void* gaddr) {
    unsigned long long ga = (unsigned long long)(uintptr_t)gaddr;
    v4u g0;
    g0.x = 1u;                                                  // count=1, user mode
    g0.y = lds_off;                                             // lds_addr (bytes)
    g0.z = (unsigned)ga;                                        // global_addr[31:0]
    g0.w = (unsigned)((ga >> 32) & 0x01FFFFFFu) | (2u << 30);   // global_addr[56:32] | type=2
    v8u g1;
    g1.s0 = (1u << 16)            // data_size = 2 bytes
          | (1u << 20)            // pad_enable
          | (4u << 22)            // pad_interval: 32 DWORDs (= one 64-elem bf16 row)
          | (3u << 25);           // pad_amount: 4 DWORDs (= 8 bf16 of padding)
    g1.s1 = (64u << 16);          // tensor_dim0 = 64 (tile == tensor, no OOB)
    g1.s2 = (128u << 16);         // tensor_dim1 = 128
    g1.s3 = (64u << 16);          // tile_dim0 = 64
    g1.s4 = 128u;                 // tile_dim1 = 128, tile_dim2 = 0
    g1.s5 = 512u;                 // tensor_dim0_stride[31:0] = 512 elements
    g1.s6 = 0u;
    g1.s7 = 0u;
    asm volatile("tensor_load_to_lds %0, %1" :: "s"(g0), "s"(g1) : "memory");
}

// ---------------- Kernel A: normalize x rows, split to bf16 hi/lo ----------------
__global__ __launch_bounds__(256) void k_prep_x(const float* __restrict__ x,
                                                unsigned short* __restrict__ xh,
                                                unsigned short* __restrict__ xl,
                                                float* __restrict__ rnx) {
    const int n = blockIdx.x, tid = threadIdx.x;
    __shared__ float red[256];
    const float* row = x + (size_t)n * DDIM;
    float v0 = row[tid], v1 = row[tid + 256];
    red[tid] = v0 * v0 + v1 * v1;
    __syncthreads();
    for (int o = 128; o > 0; o >>= 1) { if (tid < o) red[tid] += red[tid + o]; __syncthreads(); }
    float rs = 1.0f / sqrtf(red[0]);
    if (tid == 0) rnx[n] = rs;
    __bf16 h, l;
    float a0 = v0 * rs, a1 = v1 * rs;
    splitbf(a0, h, l); xh[(size_t)n * DDIM + tid]       = bfbits(h); xl[(size_t)n * DDIM + tid]       = bfbits(l);
    splitbf(a1, h, l); xh[(size_t)n * DDIM + tid + 256] = bfbits(h); xl[(size_t)n * DDIM + tid + 256] = bfbits(l);
}

// ---------------- Kernel W: per-class 1/||w_c|| (also warms L2 for the GEMM) ----------------
__global__ __launch_bounds__(256) void k_wnorm(const float* __restrict__ w, float* __restrict__ wni) {
    const int wv = threadIdx.x >> 5, lane = threadIdx.x & 31;
    const int c = blockIdx.x * 8 + wv;
    if (c >= NCLS) return;
    const float4* p = (const float4*)(w + (size_t)c * DDIM) + lane * 4;
    float s = 0.f;
#pragma unroll
    for (int i = 0; i < 4; ++i) { float4 v = p[i]; s += v.x * v.x + v.y * v.y + v.z * v.z + v.w * v.w; }
#pragma unroll
    for (int o = 16; o > 0; o >>= 1) s += __shfl_xor(s, o, 32);
    if (lane == 0) wni[c] = 1.0f / sqrtf(s);
}

// ---------------- Kernel B: exact f32 target logit per row ----------------
__global__ __launch_bounds__(256) void k_tlogit(const float* __restrict__ x, const float* __restrict__ w,
                                                const int* __restrict__ tgt, const float* __restrict__ rnx,
                                                float* __restrict__ tl) {
    const int n = blockIdx.x, tid = threadIdx.x;
    const int c = tgt[n];
    __shared__ float rd[256], rw[256];
    const float* xr = x + (size_t)n * DDIM;
    const float* wr = w + (size_t)c * DDIM;
    float d = 0.f, q = 0.f;
    for (int k = tid; k < DDIM; k += 256) { float a = xr[k], b = wr[k]; d += a * b; q += b * b; }
    rd[tid] = d; rw[tid] = q; __syncthreads();
    for (int o = 128; o > 0; o >>= 1) {
        if (tid < o) { rd[tid] += rd[tid + o]; rw[tid] += rw[tid + o]; }
        __syncthreads();
    }
    if (tid == 0) {
        float v = rd[0] * rnx[n] / sqrtf(rw[0]);
        v = fminf(fmaxf(v, -1.0f + EPSC), 1.0f - EPSC);
        tl[n] = v;
    }
}

// ---------------- Kernel C: t_new EMA, cos(theta+m), final target logit; zero the output ----------------
__global__ __launch_bounds__(512) void k_scalar(const float* __restrict__ tl, const float* __restrict__ t_in,
                                                float* __restrict__ tnew, float* __restrict__ cosm,
                                                float* __restrict__ ftl, float* __restrict__ out) {
    const int tid = threadIdx.x;
    __shared__ float rd[512];
    float v = tl[tid];
    rd[tid] = v; __syncthreads();
    for (int o = 256; o > 0; o >>= 1) { if (tid < o) rd[tid] += rd[tid + o]; __syncthreads(); }
    if (tid == 0) {
        float mean = rd[0] * (1.0f / NROWS);
        tnew[0] = mean * 0.01f + 0.99f * t_in[0];
        out[0] = 0.0f;
    }
    float sn = sqrtf(fmaxf(0.f, 1.0f - v * v));
    float cm = v * COS_M_F - sn * SIN_M_F;
    cosm[tid] = cm;
    ftl[tid] = (v > THRESH_F) ? cm : (v - MM_F);
}

// ---------------- Kernel D: split-bf16 WMMA GEMM, TDM-staged A, fused epilogue + online logsumexp ----------------
struct alignas(16) SMem {
    __bf16 Ah[128][72];
    __bf16 Al[128][72];
    __bf16 Bh[128][72];
    __bf16 Bl[128][72];
    float wns[128], cms[128], fts[128];
    int   tgs[128];
    float rm[4][128], rsum[4][128];
};

__global__ __launch_bounds__(256) void k_gemm(const unsigned short* __restrict__ xh_u,
                                              const unsigned short* __restrict__ xl_u,
                                              const float* __restrict__ w,
                                              const float* __restrict__ wni,
                                              const int* __restrict__ tgt,
                                              const float* __restrict__ cosm,
                                              const float* __restrict__ ftl,
                                              const float* __restrict__ tnewp,
                                              float* __restrict__ Pm, float* __restrict__ Ps) {
    __shared__ SMem sm;

    const int tid = threadIdx.x;
    const int bx = blockIdx.x, by = blockIdx.y;
    const int wv = tid >> 5, lane = tid & 31;
    const int wm = wv >> 2, wn = wv & 3;       // 2x4 wave grid: 64 rows x 32 cols per wave
    const int half = lane >> 4, mh = lane & 15;

    if (tid < 128) {
        int gr = by * 128 + tid;
        sm.tgs[tid] = tgt[gr]; sm.cms[tid] = cosm[gr]; sm.fts[tid] = ftl[gr];
    } else {
        int cc = tid - 128, gc0 = bx * 128 + cc;
        sm.wns[cc] = (gc0 < NCLS) ? wni[gc0] : 0.f;
    }
    const float tnew = tnewp[0];

    const unsigned ldsAh = (unsigned)(uintptr_t)&sm.Ah[0][0];  // LDS aperture: addr[31:0] = LDS offset
    const unsigned ldsAl = (unsigned)(uintptr_t)&sm.Al[0][0];
    const size_t abase = (size_t)(by * 128) * DDIM;

    v8f acc[4][2];
#pragma unroll
    for (int i = 0; i < 4; ++i)
#pragma unroll
        for (int j = 0; j < 2; ++j)
#pragma unroll
            for (int e = 0; e < 8; ++e) acc[i][j][e] = 0.f;

    const int lc = tid >> 1;           // 0..127 : W tile row (class)
    const int ko = (tid & 1) * 32;     // 0 or 32 within K-step
    const int gc = bx * 128 + lc;
    const size_t wrow = (size_t)gc * DDIM;

    for (int k0 = 0; k0 < DDIM; k0 += 64) {
        // ---- TDM: async-stage A hi/lo tiles (bf16) into padded LDS ----
        if (wv == 0) {
            tdm_load_a_tile(ldsAh, xh_u + abase + k0);
            tdm_load_a_tile(ldsAl, xl_u + abase + k0);
        }
        // ---- stage W tile (f32 -> bf16 hi/lo) through VGPRs ----
        if (gc < NCLS) {
            const float4* src = (const float4*)(w + wrow + k0 + ko);
#pragma unroll
            for (int i = 0; i < 4; ++i) {
                float4 f0 = src[2 * i], f1 = src[2 * i + 1];
                float e[8] = { f0.x, f0.y, f0.z, f0.w, f1.x, f1.y, f1.z, f1.w };
                unsigned hu[4], lu[4];
#pragma unroll
                for (int j = 0; j < 4; ++j) {
                    __bf16 h0, l0, h1, l1;
                    splitbf(e[2 * j], h0, l0); splitbf(e[2 * j + 1], h1, l1);
                    hu[j] = pack2(h0, h1); lu[j] = pack2(l0, l1);
                }
                *(uint4*)&sm.Bh[lc][ko + i * 8] = make_uint4(hu[0], hu[1], hu[2], hu[3]);
                *(uint4*)&sm.Bl[lc][ko + i * 8] = make_uint4(lu[0], lu[1], lu[2], lu[3]);
            }
            if (k0 + 64 < DDIM) __builtin_prefetch(w + wrow + k0 + 64 + ko, 0, 1);
        } else {
            uint4 z = make_uint4(0u, 0u, 0u, 0u);
#pragma unroll
            for (int i = 0; i < 4; ++i) {
                *(uint4*)&sm.Bh[lc][ko + i * 8] = z;
                *(uint4*)&sm.Bl[lc][ko + i * 8] = z;
            }
        }
        // ---- publish: issuing wave drains TENSORcnt, then workgroup barrier ----
        if (wv == 0) __builtin_amdgcn_s_wait_tensorcnt(0);
        __syncthreads();

#pragma unroll
        for (int kc = 0; kc < 64; kc += 32) {
            v16bf ah[4], al[4], bh[2], bl[2];
#pragma unroll
            for (int mt = 0; mt < 4; ++mt) {
                const int r = wm * 64 + mt * 16 + mh;
                ah[mt] = ldA(&sm.Ah[r][0], kc, half);
                al[mt] = ldA(&sm.Al[r][0], kc, half);
            }
#pragma unroll
            for (int nt = 0; nt < 2; ++nt) {
                const int c = wn * 32 + nt * 16 + mh;
                bh[nt] = ldB(&sm.Bh[c][0], kc, half);
                bl[nt] = ldB(&sm.Bl[c][0], kc, half);
            }
#pragma unroll
            for (int mt = 0; mt < 4; ++mt)
#pragma unroll
                for (int nt = 0; nt < 2; ++nt) {
                    acc[mt][nt] = __builtin_amdgcn_wmma_f32_16x16x32_bf16(
                        false, ah[mt], false, bh[nt], (short)0, acc[mt][nt], false, false);
                    acc[mt][nt] = __builtin_amdgcn_wmma_f32_16x16x32_bf16(
                        false, ah[mt], false, bl[nt], (short)0, acc[mt][nt], false, false);
                    acc[mt][nt] = __builtin_amdgcn_wmma_f32_16x16x32_bf16(
                        false, al[mt], false, bh[nt], (short)0, acc[mt][nt], false, false);
                }
        }
        __syncthreads();
    }

    // ---- fused epilogue: margin ops + online (max,sumexp) per row over this 128-col block ----
#pragma unroll
    for (int mt = 0; mt < 4; ++mt) {
#pragma unroll
        for (int j = 0; j < 8; ++j) {
            const int rl = wm * 64 + mt * 16 + half * 8 + j;   // C/D layout: VGPR j -> row j + 8*half
            const float cm = sm.cms[rl], ft = sm.fts[rl];
            const int tg = sm.tgs[rl];
            float lg[2];
#pragma unroll
            for (int nt = 0; nt < 2; ++nt) {
                const int cl = wn * 32 + nt * 16 + mh;
                const int gcl = bx * 128 + cl;
                float cosv = acc[mt][nt][j] * sm.wns[cl];
                cosv = fminf(fmaxf(cosv, -1.0f + EPSC), 1.0f - EPSC);
                if (cosv > cm) cosv = cosv * (tnew + cosv);    // hard-example reweight
                if (gcl == tg) cosv = ft;                      // scatter margin-adjusted logit
                lg[nt] = (gcl < NCLS) ? SCALEF * cosv : -1e30f;
            }
            float m2 = fmaxf(lg[0], lg[1]);
            float s2 = expf(lg[0] - m2) + expf(lg[1] - m2);
#pragma unroll
            for (int o = 1; o < 16; o <<= 1) {
                float mo = __shfl_xor(m2, o, 32);
                float so = __shfl_xor(s2, o, 32);
                float mn = fmaxf(m2, mo);
                s2 = s2 * expf(m2 - mn) + so * expf(mo - mn);
                m2 = mn;
            }
            if (mh == 0) { sm.rm[wn][rl] = m2; sm.rsum[wn][rl] = s2; }
        }
    }
    __syncthreads();
    if (tid < 128) {
        float M = sm.rm[0][tid], S = sm.rsum[0][tid];
#pragma unroll
        for (int wq = 1; wq < 4; ++wq) {
            float mo = sm.rm[wq][tid], so = sm.rsum[wq][tid];
            float mn = fmaxf(M, mo);
            S = S * expf(M - mn) + so * expf(mo - mn);
            M = mn;
        }
        const size_t idx = (size_t)bx * NROWS + by * 128 + tid;
        Pm[idx] = M; Ps[idx] = S;
    }
}

// ---------------- Kernel E: combine partials -> logsumexp -> mean loss ----------------
__global__ __launch_bounds__(128) void k_lse(const float* __restrict__ Pm, const float* __restrict__ Ps,
                                             const float* __restrict__ ftl, float* __restrict__ out) {
    const int n = blockIdx.x, tid = threadIdx.x;
    float M = -1e30f, S = 0.f;
    for (int i = tid; i < NBX; i += 128) {
        float mo = Pm[(size_t)i * NROWS + n], so = Ps[(size_t)i * NROWS + n];
        float mn = fmaxf(M, mo);
        S = S * expf(M - mn) + so * expf(mo - mn);
        M = mn;
    }
    __shared__ float smax[128], ssum[128];
    smax[tid] = M; ssum[tid] = S; __syncthreads();
    for (int o = 64; o > 0; o >>= 1) {
        if (tid < o) {
            float mo = smax[tid + o], so = ssum[tid + o];
            float mn = fmaxf(smax[tid], mo);
            ssum[tid] = ssum[tid] * expf(smax[tid] - mn) + so * expf(mo - mn);
            smax[tid] = mn;
        }
        __syncthreads();
    }
    if (tid == 0) {
        float lse = smax[0] + logf(ssum[0]);
        atomicAdd(out, (lse - SCALEF * ftl[n]) * (1.0f / NROWS));
    }
}

extern "C" void kernel_launch(void* const* d_in, const int* in_sizes, int n_in,
                              void* d_out, int out_size, void* d_ws, size_t ws_size,
                              hipStream_t stream) {
    (void)in_sizes; (void)n_in; (void)out_size; (void)ws_size;
    const float* x   = (const float*)d_in[0];
    const float* w   = (const float*)d_in[1];
    const float* t   = (const float*)d_in[2];
    const int*   tgt = (const int*)d_in[3];
    float* out = (float*)d_out;

    char* p = (char*)d_ws;
    auto alloc = [&](size_t bytes) -> char* {
        char* r = p; p += (bytes + 255) & ~(size_t)255; return r;
    };
    unsigned short* xh  = (unsigned short*)alloc((size_t)NROWS * DDIM * 2);
    unsigned short* xl  = (unsigned short*)alloc((size_t)NROWS * DDIM * 2);
    float* rnx  = (float*)alloc(NROWS * 4);
    float* tl   = (float*)alloc(NROWS * 4);
    float* cosm = (float*)alloc(NROWS * 4);
    float* ftl  = (float*)alloc(NROWS * 4);
    float* tnew = (float*)alloc(256);
    float* wni  = (float*)alloc((size_t)NCLS * 4);
    float* Pm   = (float*)alloc((size_t)NBX * NROWS * 4);
    float* Ps   = (float*)alloc((size_t)NBX * NROWS * 4);

    k_prep_x <<<NROWS, 256, 0, stream>>>(x, xh, xl, rnx);
    k_wnorm  <<<(NCLS + 7) / 8, 256, 0, stream>>>(w, wni);
    k_tlogit <<<NROWS, 256, 0, stream>>>(x, w, tgt, rnx, tl);
    k_scalar <<<1, 512, 0, stream>>>(tl, t, tnew, cosm, ftl, out);
    k_gemm   <<<dim3(NBX, NROWS / 128), 256, 0, stream>>>(xh, xl, w, wni, tgt, cosm, ftl, tnew, Pm, Ps);
    k_lse    <<<NROWS, 128, 0, stream>>>(Pm, Ps, ftl, out);
}
